// Amodel_20933670600894
// MI455X (gfx1250) — compile-verified, matrix operand order checked by reading, not run on hardware
//
#include <hip/hip_runtime.h>

typedef __attribute__((ext_vector_type(16))) _Float16 v16h;
typedef __attribute__((ext_vector_type(8)))  float    v8f;

#define BN 256
#define TT 1024
#define SDIM 64
#define FDIM 128
#define HH 128
#define G3 384

// K index mapping inside a 16-bit A/B WMMA fragment (per CDNA5 ISA 7.12.2):
// lanes 0-15 hold K {0..7,16..23}, lanes 16-31 hold K {8..15,24..31}, paired per VGPR.
static __device__ __forceinline__ int kmap(int e) {
  int v = e >> 1, c = e & 1;
  return (v < 4) ? (2 * v + c) : (16 + 2 * (v - 4) + c);
}

// A fragment: 16x32 tile from row-major f16 src (leading dim ld), origin (r0,k0)
static __device__ __forceinline__ v16h load_frag_a(const _Float16* src, int ld,
                                                   int r0, int k0, int lane) {
  v16h a;
  int row = r0 + (lane & 15);
  int kb = (lane < 16) ? 0 : 8;
#pragma unroll
  for (int e = 0; e < 16; ++e) a[e] = src[row * ld + k0 + kb + kmap(e)];
  return a;
}

// B fragment: B[k][n] = W[n][k] with W row-major f16 (leading dim ldw), origin (k0,n0)
static __device__ __forceinline__ v16h load_frag_b(const _Float16* W, int ldw,
                                                   int k0, int n0, int lane) {
  v16h b;
  int n = n0 + (lane & 15);
  int kb = (lane < 16) ? 0 : 8;
#pragma unroll
  for (int e = 0; e < 16; ++e) b[e] = W[n * ldw + k0 + kb + kmap(e)];
  return b;
}

static __device__ __forceinline__ float sigmoidf_(float x) {
  return 1.0f / (1.0f + __expf(-x));
}
static __device__ __forceinline__ float leakyf_(float x) {
  return (x >= 0.0f) ? x : 0.01f * x;
}

// ---- CDNA5 async memory->LDS copy (ASYNCcnt-tracked, no VGPR round trip) ----
// GVS addressing: mem = SGPR_base + VGPR_i32_offset; LDS dest offset in VGPR.
static __device__ __forceinline__ void async_load_b128(unsigned lds_off,
                                                       unsigned voff,
                                                       const void* sbase) {
  asm volatile("global_load_async_to_lds_b128 %0, %1, %2 offset:0"
               :
               : "v"(lds_off), "v"(voff), "s"(sbase)
               : "memory");
}
// async loads complete in order: waiting for <=1 retires the older (current)
// tile while the just-issued (next) tile stays in flight.
static __device__ __forceinline__ void wait_async_le1() {
  asm volatile("s_wait_asynccnt 0x1" ::: "memory");
}
static __device__ __forceinline__ void wait_async_le0() {
  asm volatile("s_wait_asynccnt 0x0" ::: "memory");
}

// ---------------- kernel 0: weight f32 -> f16 conversion ----------------
__global__ void k_prep(const float* w_in, const float* wi_f, const float* wh_f,
                       _Float16* w_in_h, _Float16* wi_f_h, _Float16* wh_f_h) {
  int i = blockIdx.x * 256 + threadIdx.x;
  if (i < HH * SDIM) w_in_h[i] = (_Float16)w_in[i];
  if (i < G3 * HH) {
    wi_f_h[i] = (_Float16)wi_f[i];
    wh_f_h[i] = (_Float16)wh_f[i];
  }
}

// ---------------- kernel 1: lengths from mask ----------------
__global__ void k_len(const float* mask, int* len) {
  __shared__ float red[256];
  int b = blockIdx.x;
  float s = 0.f;
  for (int t = threadIdx.x; t < TT; t += 256) s += mask[(long)b * TT + t];
  red[threadIdx.x] = s;
  __syncthreads();
  for (int o = 128; o > 0; o >>= 1) {
    if (threadIdx.x < o) red[threadIdx.x] += red[threadIdx.x + o];
    __syncthreads();
  }
  if (threadIdx.x == 0) len[b] = (int)(red[0] + 0.5f);
}

// ------- kernel 2: fused input projection + LayerNorm + gate GEMM (WMMA) -------
__global__ __launch_bounds__(256) void k_proj_gates(
    const float* __restrict__ series, const float* __restrict__ b_in,
    const float* __restrict__ ln_g, const float* __restrict__ ln_b,
    const float* __restrict__ gi_bias, const _Float16* __restrict__ w_in_h,
    const _Float16* __restrict__ wi_f_h, const int* __restrict__ len,
    float* __restrict__ xlast, _Float16* __restrict__ gates) {
  __shared__ _Float16 a0h[64][SDIM];    // 8 KB  series tile (f16)
  __shared__ float    x1s[64][HH];      // 32 KB pre-LN x1
  __shared__ _Float16 x1h[64][HH];      // 16 KB post-LN x1 (f16)
  __shared__ float red_s[256], red_q[256];
  __shared__ float mu_s[64], rs_s[64];

  const int tid = threadIdx.x;
  const int wave = tid >> 5, lane = tid & 31;
  const long row0 = (long)blockIdx.x * 64;     // flat (b,t) row
  const int b = (int)(row0 / TT);
  const int t0 = (int)(row0 % TT);

  // stage series tile [64][64] as f16
  for (int i = tid; i < 64 * SDIM; i += 256) {
    int r = i >> 6, c = i & 63;
    a0h[r][c] = (_Float16)series[(row0 + r) * SDIM + c];
  }
  __syncthreads();

  // GEMM1: x1 = series @ w_in^T   (4 M-tiles x 8 N-tiles, 4 tiles per wave)
  for (int j = 0; j < 4; ++j) {
    int tIdx = wave * 4 + j;
    int mi = tIdx >> 3, ni = tIdx & 7;
    v8f acc = {};
#pragma unroll
    for (int kb = 0; kb < 2; ++kb) {
      v16h af = load_frag_a(&a0h[0][0], SDIM, mi * 16, kb * 32, lane);
      v16h bf = load_frag_b(w_in_h, SDIM, kb * 32, ni * 16, lane);
      acc = __builtin_amdgcn_wmma_f32_16x16x32_f16(false, af, false, bf,
                                                   (short)0, acc, false, false);
    }
    int n = ni * 16 + (lane & 15);
    float bias = b_in[n];
#pragma unroll
    for (int r = 0; r < 8; ++r) {
      int m = mi * 16 + r + ((lane < 16) ? 0 : 8);
      x1s[m][n] = acc[r] + bias;
    }
  }
  __syncthreads();

  // LayerNorm over last dim (128); 4 threads per row
  {
    int r = tid >> 2, part = tid & 3;
    float s = 0.f, q = 0.f;
    for (int c = part * 32; c < part * 32 + 32; ++c) {
      float v = x1s[r][c];
      s += v;
      q += v * v;
    }
    red_s[tid] = s;
    red_q[tid] = q;
  }
  __syncthreads();
  if ((tid & 3) == 0) {
    int r = tid >> 2;
    float s = red_s[tid] + red_s[tid + 1] + red_s[tid + 2] + red_s[tid + 3];
    float q = red_q[tid] + red_q[tid + 1] + red_q[tid + 2] + red_q[tid + 3];
    float mu = s * (1.0f / HH);
    float var = q * (1.0f / HH) - mu * mu;
    mu_s[r] = mu;
    rs_s[r] = rsqrtf(var + 1e-5f);
  }
  __syncthreads();
  {
    int lenb = len[b];
    for (int i = tid; i < 64 * HH; i += 256) {
      int r = i >> 7, c = i & 127;
      float v = (x1s[r][c] - mu_s[r]) * rs_s[r] * ln_g[c] + ln_b[c];
      x1h[r][c] = (_Float16)v;
      if (t0 + r == lenb - 1) xlast[b * HH + c] = v;  // row needed by bwd cell
    }
  }
  __syncthreads();

  // GEMM2: gates = x1 @ gru_wi_f^T + bi  (4 x 24 tiles, 12 per wave), store f16
  for (int j = 0; j < 12; ++j) {
    int tIdx = wave * 12 + j;
    int mi = tIdx / 24, ni = tIdx % 24;
    v8f acc = {};
#pragma unroll
    for (int kb = 0; kb < 4; ++kb) {
      v16h af = load_frag_a(&x1h[0][0], HH, mi * 16, kb * 32, lane);
      v16h bf = load_frag_b(wi_f_h, HH, kb * 32, ni * 16, lane);
      acc = __builtin_amdgcn_wmma_f32_16x16x32_f16(false, af, false, bf,
                                                   (short)0, acc, false, false);
    }
    int n = ni * 16 + (lane & 15);
    float bias = gi_bias[n];
#pragma unroll
    for (int r = 0; r < 8; ++r) {
      int m = mi * 16 + r + ((lane < 16) ? 0 : 8);
      gates[(row0 + m) * G3 + n] = (_Float16)(acc[r] + bias);
    }
  }
}

// ------- kernel 3: persistent GRU scan (16 WGs, 16 batch rows each) -------
// Gate tiles stream memory->LDS via ASYNCcnt-tracked async copies, double
// buffered, so the staging never passes through VGPRs on the critical path.
__global__ __launch_bounds__(256) void k_scan(
    const _Float16* __restrict__ gates, const float* __restrict__ mask,
    const float* __restrict__ bh, const _Float16* __restrict__ wh_h,
    float* __restrict__ h_out) {
  __shared__ float    ghs[16][G3];        // 24 KB  h @ wh^T
  __shared__ _Float16 gx[2][16][G3];      // 24 KB  double-buffered gate tiles
  __shared__ float    hf[16][HH];         // 8 KB   h (f32)
  __shared__ _Float16 hh[16][HH];         // 4 KB   h (f16, WMMA A source)
  __shared__ float    bhs[G3];            // 1.5 KB
  __shared__ float    ms[16];             // per-step mask values

  const int tid = threadIdx.x;
  const int wave = tid >> 5, lane = tid & 31;
  const int b0 = blockIdx.x * 16;

  for (int i = tid; i < G3; i += 256) bhs[i] = bh[i];
  for (int i = tid; i < 16 * HH; i += 256) {
    hf[i >> 7][i & 127] = 0.f;
    hh[i >> 7][i & 127] = (_Float16)0.f;
  }

  // recurrent weight B-fragments hoisted into registers (loop-invariant):
  // wave owns N-tiles wave*3 .. wave*3+2, each with 4 K-steps -> 96 VGPRs
  v16h bw[3][4];
#pragma unroll
  for (int nj = 0; nj < 3; ++nj)
#pragma unroll
    for (int kb = 0; kb < 4; ++kb)
      bw[nj][kb] = load_frag_b(wh_h, HH, kb * 32, (wave * 3 + nj) * 16, lane);

  // async staging parameters: 16 rows x 768 B per tile, one b128 per thread
  const int arow = tid / 12, avec = tid % 12;  // valid for tid < 192
  const unsigned lds_base0 =
      (unsigned)(uintptr_t)&gx[0][0][0];  // LDS aperture: low 32b = LDS offset
  const unsigned lds_base1 = (unsigned)(uintptr_t)&gx[1][0][0];

  // preload gate tile for t = 0 into buffer 0
  if (tid < 192) {
    unsigned voff =
        (unsigned)((((long)(b0 + arow) * TT + 0) * G3) * 2 + avec * 16);
    async_load_b128(lds_base0 + (unsigned)(arow * G3 * 2 + avec * 16), voff,
                    gates);
  }
  __syncthreads();

  int cur = 0;
  for (int t = 0; t < TT; ++t) {
    // gh = h @ wh^T via WMMA (1 M-tile x 24 N-tiles; 3 per wave)
    v16h af[4];
#pragma unroll
    for (int kb = 0; kb < 4; ++kb)
      af[kb] = load_frag_a(&hh[0][0], HH, 0, kb * 32, lane);
#pragma unroll
    for (int nj = 0; nj < 3; ++nj) {
      v8f acc = {};
#pragma unroll
      for (int kb = 0; kb < 4; ++kb)
        acc = __builtin_amdgcn_wmma_f32_16x16x32_f16(
            false, af[kb], false, bw[nj][kb], (short)0, acc, false, false);
      int n = (wave * 3 + nj) * 16 + (lane & 15);
#pragma unroll
      for (int r = 0; r < 8; ++r) ghs[r + ((lane < 16) ? 0 : 8)][n] = acc[r];
    }

    // stage this step's 16 mask values once
    if (tid < 16) ms[tid] = mask[(long)(b0 + tid) * TT + t];

    // kick async copy of the t+1 gate tile into the other buffer
    if (t + 1 < TT) {
      if (tid < 192) {
        unsigned ldsb = (cur == 0) ? lds_base1 : lds_base0;
        unsigned voff = (unsigned)((((long)(b0 + arow) * TT + (t + 1)) * G3) * 2 +
                                   avec * 16);
        async_load_b128(ldsb + (unsigned)(arow * G3 * 2 + avec * 16), voff,
                        gates);
        if (t + 2 < TT)
          __builtin_prefetch(&gates[((long)(b0 + arow) * TT + (t + 2)) * G3], 0,
                             1);
      }
      wait_async_le1();  // in-order completion: current tile is resident
    } else {
      wait_async_le0();
    }
    __syncthreads();

    // fused GRU activations + mask freeze (8 elements per thread)
#pragma unroll
    for (int i = 0; i < 8; ++i) {
      int e = tid + i * 256;
      int r = e >> 7, j = e & 127;
      float gr = (float)gx[cur][r][j] + ghs[r][j] + bhs[j];
      float gz = (float)gx[cur][r][HH + j] + ghs[r][HH + j] + bhs[HH + j];
      float gnx = (float)gx[cur][r][2 * HH + j];
      float ghn = ghs[r][2 * HH + j] + bhs[2 * HH + j];
      float rr = sigmoidf_(gr);
      float zz = sigmoidf_(gz);
      float nn = tanhf(gnx + rr * ghn);
      float hold = hf[r][j];
      float hnew = (1.f - zz) * nn + zz * hold;
      hnew = (ms[r] > 0.f) ? hnew : hold;
      hf[r][j] = hnew;
      hh[r][j] = (_Float16)hnew;
    }
    __syncthreads();
    cur ^= 1;
  }
  for (int i = tid; i < 16 * HH; i += 256)
    h_out[(long)b0 * HH + i] = hf[i >> 7][i & 127];
}

// ------- kernel 4: backward cell + feature MLP + fusion head -------
__global__ __launch_bounds__(256) void k_tail(
    const float* __restrict__ hfwd, const float* __restrict__ xlast,
    const float* __restrict__ wi_b, const float* __restrict__ bi_b,
    const float* __restrict__ bh_b, const float* __restrict__ feature,
    const float* __restrict__ fw0, const float* __restrict__ fb0,
    const float* __restrict__ bn0g, const float* __restrict__ bn0b,
    const float* __restrict__ hidw, const float* __restrict__ hidb,
    const float* __restrict__ hbng, const float* __restrict__ hbnb,
    const float* __restrict__ w1, const float* __restrict__ b1,
    const float* __restrict__ w2, const float* __restrict__ b2,
    const float* __restrict__ w3, const float* __restrict__ b3,
    float* __restrict__ out) {
  __shared__ float pool[16][2 * HH];  // 16 KB  [h_fwd | h_bwd]
  __shared__ float tmp[16][G3];       // 24 KB  gx_b / concat
  __shared__ float va[16][HH];        // 8 KB
  __shared__ float vb[16][HH];        // 8 KB
  const int tid = threadIdx.x;
  const int b0 = blockIdx.x * 16;
  const float bns = rsqrtf(1.f + 1e-5f);  // eval BatchNorm scale

  for (int i = tid; i < 16 * HH; i += 256) {
    pool[i >> 7][i & 127] = hfwd[(long)b0 * HH + i];
    va[i >> 7][i & 127] = xlast[(long)b0 * HH + i];
  }
  __syncthreads();
  // gx_b = x_last @ wi_b^T + bi_b
  for (int i = tid; i < 16 * G3; i += 256) {
    int r = i / G3, n = i % G3;
    float s = bi_b[n];
    for (int k = 0; k < HH; ++k) s += va[r][k] * wi_b[n * HH + k];
    tmp[r][n] = s;
  }
  __syncthreads();
  // backward cell with h0 = 0  ->  gh == bh_b ;  h_bwd = (1-z)*n
  for (int i = tid; i < 16 * HH; i += 256) {
    int r = i >> 7, j = i & 127;
    float rr = sigmoidf_(tmp[r][j] + bh_b[j]);
    float zz = sigmoidf_(tmp[r][HH + j] + bh_b[HH + j]);
    float nn = tanhf(tmp[r][2 * HH + j] + rr * bh_b[2 * HH + j]);
    pool[r][HH + j] = (1.f - zz) * nn;
    va[r][j] = feature[(long)(b0 + r) * FDIM + j];  // stage feature rows
  }
  __syncthreads();
  // feature layer 0 (Linear + eval-BN + leaky)
  for (int i = tid; i < 16 * HH; i += 256) {
    int r = i >> 7, j = i & 127;
    float s = fb0[j];
    for (int k = 0; k < FDIM; ++k) s += va[r][k] * fw0[j * FDIM + k];
    vb[r][j] = leakyf_(s * bns * bn0g[j] + bn0b[j]);
  }
  __syncthreads();
  // two hidden layers, ping-pong vb -> va -> vb
  for (int L = 0; L < 2; ++L) {
    float(*src)[HH] = (L == 0) ? vb : va;
    float(*dst)[HH] = (L == 0) ? va : vb;
    for (int i = tid; i < 16 * HH; i += 256) {
      int r = i >> 7, j = i & 127;
      float s = hidb[L * HH + j];
      for (int k = 0; k < HH; ++k) s += src[r][k] * hidw[(L * HH + j) * HH + k];
      dst[r][j] = leakyf_(s * bns * hbng[L * HH + j] + hbnb[L * HH + j]);
    }
    __syncthreads();
  }
  // concat [pool(256) | x2(128 in vb)] into tmp
  for (int i = tid; i < 16 * G3; i += 256) {
    int r = i / G3, c = i % G3;
    tmp[r][c] = (c < 2 * HH) ? pool[r][c] : vb[r][c - 2 * HH];
  }
  __syncthreads();
  // y1 = leaky(x @ w1^T + b1)
  for (int i = tid; i < 16 * HH; i += 256) {
    int r = i >> 7, j = i & 127;
    float s = b1[j];
    for (int k = 0; k < G3; ++k) s += tmp[r][k] * w1[j * G3 + k];
    va[r][j] = leakyf_(s);
  }
  __syncthreads();
  // y2 = leaky(y1 @ w2^T + b2)
  for (int i = tid; i < 16 * HH; i += 256) {
    int r = i >> 7, j = i & 127;
    float s = b2[j];
    for (int k = 0; k < HH; ++k) s += va[r][k] * w2[j * HH + k];
    vb[r][j] = leakyf_(s);
  }
  __syncthreads();
  if (tid < 16) {
    float s = b3[0];
    for (int k = 0; k < HH; ++k) s += vb[tid][k] * w3[k];
    out[b0 + tid] = sigmoidf_(s);
  }
}

extern "C" void kernel_launch(void* const* d_in, const int* in_sizes, int n_in,
                              void* d_out, int out_size, void* d_ws,
                              size_t ws_size, hipStream_t stream) {
  (void)in_sizes; (void)n_in; (void)out_size; (void)ws_size;
  const float* series  = (const float*)d_in[0];
  const float* mask    = (const float*)d_in[1];
  const float* feature = (const float*)d_in[2];
  const float* w_in    = (const float*)d_in[3];
  const float* b_in    = (const float*)d_in[4];
  const float* ln_g    = (const float*)d_in[5];
  const float* ln_b    = (const float*)d_in[6];
  const float* wi_f    = (const float*)d_in[7];
  const float* wh_f    = (const float*)d_in[8];
  const float* bi_f    = (const float*)d_in[9];
  const float* bh_f    = (const float*)d_in[10];
  const float* wi_b    = (const float*)d_in[11];
  /* d_in[12] = gru_wh_b unused: h0 == 0 for the backward single step */
  const float* bi_b    = (const float*)d_in[13];
  const float* bh_b    = (const float*)d_in[14];
  const float* fw0     = (const float*)d_in[15];
  const float* fb0     = (const float*)d_in[16];
  const float* bn0g    = (const float*)d_in[17];
  const float* bn0b    = (const float*)d_in[18];
  const float* hidw    = (const float*)d_in[19];
  const float* hidb    = (const float*)d_in[20];
  const float* hbng    = (const float*)d_in[21];
  const float* hbnb    = (const float*)d_in[22];
  const float* w1      = (const float*)d_in[23];
  const float* b1      = (const float*)d_in[24];
  const float* w2      = (const float*)d_in[25];
  const float* b2      = (const float*)d_in[26];
  const float* w3      = (const float*)d_in[27];
  const float* b3      = (const float*)d_in[28];

  // workspace layout (bytes); total ~202 MB, dominated by f16 gates
  char* ws = (char*)d_ws;
  _Float16* w_in_h = (_Float16*)(ws + 0);        //  16 KB
  _Float16* wi_f_h = (_Float16*)(ws + 16384);    //  96 KB
  _Float16* wh_f_h = (_Float16*)(ws + 114688);   //  96 KB
  int*      len    = (int*)(ws + 212992);        //   1 KB
  float*    xlast  = (float*)(ws + 214016);      // 128 KB
  float*    hfwd   = (float*)(ws + 345088);      // 128 KB
  _Float16* gates  = (_Float16*)(ws + 476160);   // 192 MB [B*T][384] f16
  float* out = (float*)d_out;

  k_prep<<<192, 256, 0, stream>>>(w_in, wi_f, wh_f, w_in_h, wi_f_h, wh_f_h);
  k_len<<<BN, 256, 0, stream>>>(mask, len);
  k_proj_gates<<<(BN * TT) / 64, 256, 0, stream>>>(
      series, b_in, ln_g, ln_b, bi_f, w_in_h, wi_f_h, len, xlast, gates);
  k_scan<<<BN / 16, 256, 0, stream>>>(gates, mask, bh_f, wh_f_h, hfwd);
  k_tail<<<BN / 16, 256, 0, stream>>>(hfwd, xlast, wi_b, bi_b, bh_b, feature,
                                      fw0, fb0, bn0g, bn0b, hidw, hidb, hbng,
                                      hbnb, w1, b1, w2, b2, w3, b3, out);
}
